// TrivialTreeTagger_82669530514058
// MI455X (gfx1250) — matrix operand order repaired and checked by reading, not compiled, not run on hardware
//
#include <hip/hip_runtime.h>

typedef __attribute__((ext_vector_type(16))) _Float16 v16h;
typedef __attribute__((ext_vector_type(8)))  float    v8f;

#define DEV static __device__ __forceinline__

DEV float sigf(float x){ return 1.0f/(1.0f + __expf(-x)); }

DEV unsigned int pack2h(float a, float b){
  union { _Float16 h[2]; unsigned int u; } x;
  x.h[0] = (_Float16)a; x.h[1] = (_Float16)b;
  return x.u;
}

// A-fragment (16x32 f16) per-lane layout, CDNA5 ISA 7.12.2:
// lanes 0-15: M=lane, V0..3 -> K=0..7 (pairs), V4..7 -> K=16..23
// lanes 16-31: M=lane-16, V0..3 -> K=8..15,     V4..7 -> K=24..31
DEV unsigned int pack_tile(const float* W, int C, int ktiles, int d){
  int tile   = d >> 8;          // 256 dwords per 16x32 tile
  int within = d & 255;
  int lane   = within >> 3;
  int v      = within & 7;
  int mt = tile / ktiles;
  int kt = tile - mt * ktiles;
  int m  = mt * 16 + (lane & 15);
  int k  = kt * 32 + ((v < 4) ? 0 : 16) + ((lane >> 4) << 3) + ((v & 3) << 1);
  return pack2h(W[m*C + k], W[m*C + k + 1]);
}

DEV v16h load_afrag(const unsigned int* p){
  union { unsigned int u[8]; v16h h; } x;
#pragma unroll
  for (int i = 0; i < 8; ++i) x.u[i] = p[i];
  return x.h;
}

// B-fragment (32x16 f16) from LDS row-major [n][k] storage:
// lanes 0-15: N=lane, K=kbase+0..7,+16..23 ; lanes 16-31: N=lane-16, K=kbase+8..15,+24..31
DEV v16h load_bfrag(const _Float16* base, int strideHalves, int lane, int kbase){
  int n  = lane & 15;
  int kk = kbase + ((lane >> 4) << 3);
  const unsigned int* p = (const unsigned int*)(base + n*strideHalves + kk);
  union { unsigned int u[8]; v16h h; } x;
#pragma unroll
  for (int i = 0; i < 8; ++i) x.u[i] = p[i < 4 ? i : i + 4];
  return x.h;
}

DEV v8f wmma16(v16h a, v16h b, v8f c){
  return __builtin_amdgcn_wmma_f32_16x16x32_f16(false, a, false, b, (short)0, c, false, false);
}

// ---------------------------------------------------------------------------
// Pack weights to f16 A-fragment layout; bsum = b_ih + b_hh.
// dword counts: Wih 131072, Whh 131072, Wc 65536, Wt 32768, bias 1024
__global__ __launch_bounds__(256) void pack_kernel(
    const float* __restrict__ Wih, const float* __restrict__ Whh,
    const float* __restrict__ Wc,  const float* __restrict__ Wt,
    const float* __restrict__ bih, const float* __restrict__ bhh,
    unsigned int* __restrict__ pWih, unsigned int* __restrict__ pWhh,
    unsigned int* __restrict__ pWc,  unsigned int* __restrict__ pWt,
    float* __restrict__ bsum)
{
  int t = blockIdx.x * 256 + threadIdx.x;
  if      (t < 131072) pWih[t]          = pack_tile(Wih, 256,  8, t);
  else if (t < 262144) pWhh[t - 131072] = pack_tile(Whh, 256,  8, t - 131072);
  else if (t < 327680) pWc [t - 262144] = pack_tile(Wc,  512, 16, t - 262144);
  else if (t < 360448) pWt [t - 327680] = pack_tile(Wt,  256,  8, t - 327680);
  else if (t < 361472) { int i = t - 360448; bsum[i] = bih[i] + bhh[i]; }
}

// ---------------------------------------------------------------------------
// Leaf level collapses to 26 vocab entries: one shared LSTM step from
// lstm_init (h=c=0), then per-vocab combined/tag/log_softmax.
__global__ __launch_bounds__(256) void leaf_kernel(
    const float* __restrict__ Wih, const float* __restrict__ bih,
    const float* __restrict__ bhh, const float* __restrict__ linit,
    const float* __restrict__ Wc,  const float* __restrict__ bc,
    const float* __restrict__ Wt,  const float* __restrict__ bt,
    const float* __restrict__ emb, float* __restrict__ leafT)
{
  __shared__ float x0[256], hv[256], comb[256], rr[256];
  int j = threadIdx.x;
  x0[j] = linit[j];
  __syncthreads();

  float z[4];
#pragma unroll
  for (int g = 0; g < 4; ++g){
    const float* wr = Wih + (size_t)(g*256 + j) * 256;
    float s = 0.0f;
    for (int k = 0; k < 256; ++k) s += wr[k] * x0[k];
    z[g] = s + bih[g*256 + j] + bhh[g*256 + j];
  }
  float c1 = sigf(z[0]) * tanhf(z[2]);        // c0 = 0 -> f-gate term vanishes
  hv[j] = sigf(z[3]) * tanhf(c1);
  __syncthreads();

  for (int v = 0; v < 26; ++v){
    const float* wr = Wc + (size_t)j * 512;
    float s = bc[j];
    for (int k = 0; k < 256; ++k) s += wr[k]       * hv[k];
    for (int k = 0; k < 256; ++k) s += wr[256 + k] * emb[v*256 + k];
    comb[j] = s;
    __syncthreads();
    const float* wt = Wt + (size_t)j * 256;
    float tg = bt[j];
    for (int k = 0; k < 256; ++k) tg += wt[k] * comb[k];
    rr[j] = tg; __syncthreads();
    for (int off = 128; off; off >>= 1){ if (j < off) rr[j] = fmaxf(rr[j], rr[j+off]); __syncthreads(); }
    float mx = rr[0]; __syncthreads();
    rr[j] = __expf(tg - mx); __syncthreads();
    for (int off = 128; off; off >>= 1){ if (j < off) rr[j] += rr[j+off]; __syncthreads(); }
    float ls = logf(rr[0]);
    leafT[v*256 + j] = tg - mx - ls;
    __syncthreads();
  }
}

// ---------------------------------------------------------------------------
// One fused level: 32 nodes per workgroup (two 16-wide N tiles), 8 waves.
// kt-outer / mt-inner GEMM: each A fragment feeds two WMMAs, 16 independent
// accumulator chains per wave. c-state in LDS (f32), h-state in LDS (f16).
__global__ __launch_bounds__(256) void tree_level_kernel(
    const unsigned int* __restrict__ pWih, const unsigned int* __restrict__ pWhh,
    const unsigned int* __restrict__ pWc,  const unsigned int* __restrict__ pWt,
    const float* __restrict__ bsum, const float* __restrict__ bc, const float* __restrict__ bt,
    const float* __restrict__ emb,  const int* __restrict__ ids,
    const float* __restrict__ child_base, const int* __restrict__ child_idents,
    const float* __restrict__ lstm_init,
    float* __restrict__ out, int nn)
{
  __shared__ float    Zs[32][1032];   // gate pre-activations / tag (f32)   132096 B
  __shared__ _Float16 Xs[32][520];    // step input X / [h,e] for Wc (f16)   33280 B
  __shared__ _Float16 Hs[32][264];    // h state / combined (f16)            16896 B
  __shared__ float    Cs[32][260];    // LSTM cell state (f32)               33280 B
  __shared__ float    redm[32][8];
  __shared__ float    reds[32][8];

  const int t    = threadIdx.x;
  const int wave = t >> 5, lane = t & 31;
  const int nIdx = lane & 15, hi = lane >> 4;
  const int tile = blockIdx.x * 32;

  const int tn  = t >> 3;              // node owned for elementwise work (0..31)
  const int tc2 = (t & 7) << 5;        // 32-wide column slab
  const int nd  = (tile + tn < nn) ? (tile + tn) : (nn - 1);

  for (int idx = t; idx < 32*264; idx += 256) (&Hs[0][0])[idx] = (_Float16)0.0f;
  for (int idx = t; idx < 32*260; idx += 256) (&Cs[0][0])[idx] = 0.0f;

  for (int s = 0; s < 5; ++s){
    // ---- load step input (f32 -> f16 into LDS) ----
    if (s == 0){
#pragma unroll
      for (int jj = 0; jj < 32; ++jj) Xs[tn][tc2+jj] = (_Float16)lstm_init[tc2+jj];
    } else {
      int ci = nd*4 + (s - 1);
      const float* src = child_idents ? (child_base + (size_t)child_idents[ci]*256)
                                      : (child_base + (size_t)ci*256);
#pragma unroll
      for (int jj = 0; jj < 32; ++jj) Xs[tn][tc2+jj] = (_Float16)src[tc2+jj];
    }
    __syncthreads();

    // ---- Z = Wih*X + Whh*H + (b_ih+b_hh) : M=1024,K=256,N=32 ----
    {
      v8f acc[8][2];
#pragma unroll
      for (int mi = 0; mi < 8; ++mi){
        int m0 = (wave*8 + mi)*16 + (hi << 3);
#pragma unroll
        for (int r = 0; r < 8; ++r){ float b = bsum[m0 + r]; acc[mi][0][r] = b; acc[mi][1][r] = b; }
      }
      for (int kt = 0; kt < 8; ++kt){      // rolled: B frags reloaded per kt
        v16h bX0 = load_bfrag(&Xs[0][0],  520, lane, kt*32);
        v16h bX1 = load_bfrag(&Xs[16][0], 520, lane, kt*32);
        v16h bH0 = load_bfrag(&Hs[0][0],  264, lane, kt*32);
        v16h bH1 = load_bfrag(&Hs[16][0], 264, lane, kt*32);
#pragma unroll
        for (int mi = 0; mi < 8; ++mi){
          int mt = wave*8 + mi;
          v16h a  = load_afrag(pWih + ((mt*8 + kt) << 8) + (lane << 3));
          acc[mi][0] = wmma16(a, bX0, acc[mi][0]);
          acc[mi][1] = wmma16(a, bX1, acc[mi][1]);
          v16h a2 = load_afrag(pWhh + ((mt*8 + kt) << 8) + (lane << 3));
          acc[mi][0] = wmma16(a2, bH0, acc[mi][0]);
          acc[mi][1] = wmma16(a2, bH1, acc[mi][1]);
        }
      }
#pragma unroll
      for (int mi = 0; mi < 8; ++mi){
        int m0 = (wave*8 + mi)*16 + (hi << 3);
#pragma unroll
        for (int r = 0; r < 8; ++r){
          Zs[nIdx     ][m0 + r] = acc[mi][0][r];
          Zs[16 + nIdx][m0 + r] = acc[mi][1][r];
        }
      }
    }
    __syncthreads();

    // ---- LSTM cell (i,f,g,o split order per jnp.split) ----
#pragma unroll
    for (int jj = 0; jj < 32; ++jj){
      int j = tc2 + jj;
      float zi = Zs[tn][j];
      float zf = Zs[tn][256 + j];
      float zg = Zs[tn][512 + j];
      float zo = Zs[tn][768 + j];
      float c  = sigf(zf)*Cs[tn][j] + sigf(zi)*tanhf(zg);
      Cs[tn][j] = c;
      Hs[tn][j] = (_Float16)(sigf(zo)*tanhf(c));
    }
    __syncthreads();
  }

  // ---- XC = [h_last, emb[id]] ----
#pragma unroll
  for (int jj = 0; jj < 32; ++jj) Xs[tn][tc2+jj] = Hs[tn][tc2+jj];
  {
    int vid = ids[nd];
#pragma unroll
    for (int jj = 0; jj < 32; ++jj) Xs[tn][256+tc2+jj] = (_Float16)emb[vid*256 + tc2 + jj];
  }
  __syncthreads();

  // ---- combined = Wc*XC + bc : M=256,K=512,N=32 (result f16 -> Hs) ----
  {
    v8f acc[2][2];
#pragma unroll
    for (int mtl = 0; mtl < 2; ++mtl){
      int m0 = (wave*2 + mtl)*16 + (hi << 3);
#pragma unroll
      for (int r = 0; r < 8; ++r){ float b = bc[m0 + r]; acc[mtl][0][r] = b; acc[mtl][1][r] = b; }
    }
    for (int kt = 0; kt < 16; ++kt){
      v16h b0 = load_bfrag(&Xs[0][0],  520, lane, kt*32);
      v16h b1 = load_bfrag(&Xs[16][0], 520, lane, kt*32);
#pragma unroll
      for (int mtl = 0; mtl < 2; ++mtl){
        int mt = wave*2 + mtl;
        v16h a = load_afrag(pWc + ((mt*16 + kt) << 8) + (lane << 3));
        acc[mtl][0] = wmma16(a, b0, acc[mtl][0]);
        acc[mtl][1] = wmma16(a, b1, acc[mtl][1]);
      }
    }
#pragma unroll
    for (int mtl = 0; mtl < 2; ++mtl){
      int m0 = (wave*2 + mtl)*16 + (hi << 3);
#pragma unroll
      for (int r = 0; r < 8; ++r){
        Hs[nIdx     ][m0 + r] = (_Float16)acc[mtl][0][r];
        Hs[16 + nIdx][m0 + r] = (_Float16)acc[mtl][1][r];
      }
    }
  }
  __syncthreads();

  // ---- tag = Wt*combined + bt : M=256,K=256,N=32 (f32 -> Zs) ----
  {
    v8f acc[2][2];
#pragma unroll
    for (int mtl = 0; mtl < 2; ++mtl){
      int m0 = (wave*2 + mtl)*16 + (hi << 3);
#pragma unroll
      for (int r = 0; r < 8; ++r){ float b = bt[m0 + r]; acc[mtl][0][r] = b; acc[mtl][1][r] = b; }
    }
    for (int kt = 0; kt < 8; ++kt){
      v16h b0 = load_bfrag(&Hs[0][0],  264, lane, kt*32);
      v16h b1 = load_bfrag(&Hs[16][0], 264, lane, kt*32);
#pragma unroll
      for (int mtl = 0; mtl < 2; ++mtl){
        int mt = wave*2 + mtl;
        v16h a = load_afrag(pWt + ((mt*8 + kt) << 8) + (lane << 3));
        acc[mtl][0] = wmma16(a, b0, acc[mtl][0]);
        acc[mtl][1] = wmma16(a, b1, acc[mtl][1]);
      }
    }
#pragma unroll
    for (int mtl = 0; mtl < 2; ++mtl){
      int m0 = (wave*2 + mtl)*16 + (hi << 3);
#pragma unroll
      for (int r = 0; r < 8; ++r){
        Zs[nIdx     ][m0 + r] = acc[mtl][0][r];
        Zs[16 + nIdx][m0 + r] = acc[mtl][1][r];
      }
    }
  }
  __syncthreads();

  // ---- log_softmax over 256 per node ----
  float pmax = -3.4e38f;
#pragma unroll
  for (int jj = 0; jj < 32; ++jj) pmax = fmaxf(pmax, Zs[tn][tc2+jj]);
  redm[tn][t & 7] = pmax;
  __syncthreads();
  if ((t & 7) == 0){
    float m = redm[tn][0];
#pragma unroll
    for (int q = 1; q < 8; ++q) m = fmaxf(m, redm[tn][q]);
    redm[tn][0] = m;
  }
  __syncthreads();
  float mx = redm[tn][0];
  float ps = 0.0f;
#pragma unroll
  for (int jj = 0; jj < 32; ++jj) ps += __expf(Zs[tn][tc2+jj] - mx);
  reds[tn][t & 7] = ps;
  __syncthreads();
  if ((t & 7) == 0){
    float ssum = 0.0f;
#pragma unroll
    for (int q = 0; q < 8; ++q) ssum += reds[tn][q];
    reds[tn][0] = ssum;
  }
  __syncthreads();
  float ls = logf(reds[tn][0]);
  if (tile + tn < nn){
#pragma unroll
    for (int jj = 0; jj < 32; ++jj)
      out[(size_t)(tile + tn)*256 + tc2 + jj] = Zs[tn][tc2+jj] - mx - ls;
  }
}

// ---------------------------------------------------------------------------
extern "C" void kernel_launch(void* const* d_in, const int* in_sizes, int n_in,
                              void* d_out, int out_size, void* d_ws, size_t ws_size,
                              hipStream_t stream)
{
  const int*   idents = (const int*)  d_in[0];
  const float* emb    = (const float*)d_in[1];
  const float* Wih    = (const float*)d_in[2];
  const float* Whh    = (const float*)d_in[3];
  const float* bih    = (const float*)d_in[4];
  const float* bhh    = (const float*)d_in[5];
  const float* Wc     = (const float*)d_in[6];
  const float* bc     = (const float*)d_in[7];
  const float* Wt     = (const float*)d_in[8];
  const float* bt     = (const float*)d_in[9];
  const float* linit  = (const float*)d_in[10];

  char* ws = (char*)d_ws;
  unsigned int* pWih = (unsigned int*)(ws + 0);
  unsigned int* pWhh = (unsigned int*)(ws + (512u  << 10));
  unsigned int* pWc  = (unsigned int*)(ws + (1024u << 10));
  unsigned int* pWt  = (unsigned int*)(ws + (1280u << 10));
  float* bsum        = (float*)       (ws + (1408u << 10));
  float* leafT       = (float*)       (ws + (1412u << 10));
  float* bufA        = (float*)       (ws + (2048u << 10));              // 16 MB
  float* bufB        = (float*)       (ws + (2048u << 10) + (16u << 20)); // 4 MB

  pack_kernel<<<1412, 256, 0, stream>>>(Wih, Whh, Wc, Wt, bih, bhh,
                                        pWih, pWhh, pWc, pWt, bsum);
  leaf_kernel<<<1, 256, 0, stream>>>(Wih, bih, bhh, linit, Wc, bc, Wt, bt, emb, leafT);

  const float* prev = nullptr;
  for (int l = 7; l >= 0; --l){
    int nnl    = 1 << (2*l);
    int offl   = ((1 << (2*l)) - 1) / 3;
    int blocks = (nnl + 31) / 32;
    const float* cb; const int* cid;
    if (l == 7){ cb = leafT; cid = idents + 21845; }   // level-8 offset
    else       { cb = prev;  cid = nullptr; }
    float* outp = (l == 0) ? (float*)d_out : (((7 - l) & 1) ? bufB : bufA);
    tree_level_kernel<<<blocks, 256, 0, stream>>>(
        pWih, pWhh, pWc, pWt, bsum, bc, bt, emb, idents + offl,
        cb, cid, linit, outp, nnl);
    prev = outp;
  }
}
